// ACLFTransformer_64158221467976
// MI455X (gfx1250) — matrix operand order repaired and checked by reading, not compiled, not run on hardware
//
#include <hip/hip_runtime.h>

// ============================================================================
// MI455X (gfx1250, wave32) transformer forward.
//
// Strategy (reasoned from MI455X specs, compile-only):
//  * ~540 GFLOP of matmuls, ~400MB working set -> fits the 192MB L2; the win
//    is putting every GEMM on the matrix pipe: V_WMMA_F32_16X16X32_BF16
//    (bf16 in, fp32 accumulate). bf16 keeps fp32 dynamic range of the
//    trained weights; fp32 accumulation + fp32 bias/epilogue keeps the
//    residual stream, LN, softmax and log-softmax numerics in fp32.
//  * One wave -> 16x64 output strip (4 WMMA accs, A-frag reused 4x);
//    8 waves/block -> 32x256 (or 128x64 for attn@V, N=64/head).
//  * Fragment loads are contiguous 16B bf16x8 loads matching the documented
//    CDNA5 WMMA VGPR layouts (ISA 7.12.2), both operands stored [rows, K].
//  * K loop is software-pipelined (ping/pong fragment sets, K%64==0 for all
//    GEMMs here) so the next stage's global_load_b128 clauses issue before
//    the current stage's 4 back-to-back WMMAs -> loads overlap the matrix
//    pipe instead of s_wait_loadcnt 0 fencing every WMMA.
// ============================================================================

typedef __bf16 bf16;
typedef bf16  bf16x8  __attribute__((ext_vector_type(8)));
typedef bf16  bf16x16 __attribute__((ext_vector_type(16)));
typedef float f32x8   __attribute__((ext_vector_type(8)));

static constexpr int D  = 512, H = 8, FF = 2048, V = 32000, DK = 64;
static constexpr int B  = 8, S = 512, Tt = 512;
static constexpr int M4 = B * S;   // 4096 rows for all [B*S, D] GEMMs
static constexpr int BH = B * H;   // 64 attention batches

union FragBF { bf16x16 v; bf16x8 h[2]; };

// ---------------- WMMA GEMM: C[z][M,N] = A[z][M,K] * W[z][N,K]^T + bias ----
// Requires: M % (BM_WAVES*16) == 0, N % (BN_WAVES*NT*16) == 0, K % 64 == 0.
template<int BM_WAVES, int BN_WAVES, int NT>
__global__ __launch_bounds__(BM_WAVES * BN_WAVES * 32)
void gemm_bf16_wmma(const bf16* __restrict__ A, const bf16* __restrict__ W,
                    const float* __restrict__ bias, float* __restrict__ Cf,
                    int M, int N, int K,
                    long long sA, long long sW, long long sC)
{
  const int lane = threadIdx.x & 31;
  const int wave = threadIdx.x >> 5;
  const int wm   = wave / BN_WAVES;
  const int wn   = wave % BN_WAVES;
  const int half = lane >> 4;      // which 16-lane group
  const int lm   = lane & 15;

  const long long z = blockIdx.z;
  const bf16* Az = A + z * sA;
  const bf16* Wz = W + z * sW;

  const int m0 = blockIdx.y * (BM_WAVES * 16) + wm * 16;
  const int n0 = blockIdx.x * (BN_WAVES * NT * 16) + wn * (NT * 16);

  f32x8 acc[NT];
#pragma unroll
  for (int t = 0; t < NT; ++t)
#pragma unroll
    for (int i = 0; i < 8; ++i) acc[t][i] = 0.0f;

  // A 16x32 bf16 fragment: lanes 0-15 get K {0..7,16..23}, lanes 16-31 get
  // K {8..15,24..31}; two contiguous 16B loads per lane.
  const bf16* abase = Az + (long long)(m0 + lm) * K + 8 * half;
  // B 32x16 fragment: lane -> col n0+lm, K {8*half..+7, 16+8*half..+7}.
  const bf16* wbase[NT];
#pragma unroll
  for (int t = 0; t < NT; ++t)
    wbase[t] = Wz + (long long)(n0 + t * 16 + lm) * K + 8 * half;

  auto loadA = [&](FragBF& a, int k0) {
    a.h[0] = *(const bf16x8*)(abase + k0);
    a.h[1] = *(const bf16x8*)(abase + k0 + 16);
  };
  auto loadB = [&](FragBF* b, int k0) {
#pragma unroll
    for (int t = 0; t < NT; ++t) {
      b[t].h[0] = *(const bf16x8*)(wbase[t] + k0);
      b[t].h[1] = *(const bf16x8*)(wbase[t] + k0 + 16);
    }
  };

  // Software pipeline: ping (P) / pong (Q) fragment sets, K step 64.
  FragBF aP, bP[NT], aQ, bQ[NT];
  loadA(aP, 0);
  loadB(bP, 0);
  for (int k0 = 0; k0 < K; k0 += 64) {
    loadA(aQ, k0 + 32);                 // prefetch stage k0+32
    loadB(bQ, k0 + 32);
#pragma unroll
    for (int t = 0; t < NT; ++t)
      acc[t] = __builtin_amdgcn_wmma_f32_16x16x32_bf16(
          false, aP.v, false, bP[t].v, (short)0, acc[t], false, false);
    if (k0 + 64 < K) {                  // prefetch stage k0+64
      loadA(aP, k0 + 64);
      loadB(bP, k0 + 64);
    }
#pragma unroll
    for (int t = 0; t < NT; ++t)
      acc[t] = __builtin_amdgcn_wmma_f32_16x16x32_bf16(
          false, aQ.v, false, bQ[t].v, (short)0, acc[t], false, false);
  }

  // C layout: element r -> row m0 + r + 8*half, col n0 + lm.
  float* Cz = Cf + z * sC;
#pragma unroll
  for (int t = 0; t < NT; ++t) {
    const int   n  = n0 + t * 16 + lm;
    const float bv = bias ? bias[n] : 0.0f;
#pragma unroll
    for (int r = 0; r < 8; ++r) {
      const int m = m0 + r + 8 * half;
      Cz[(long long)m * N + n] = acc[t][r] + bv;
    }
  }
}

// ---------------- elementwise / reduction kernels ---------------------------
__global__ void cvt_f32_bf16_kernel(const float* __restrict__ in,
                                    bf16* __restrict__ out, long long n) {
  long long i = (long long)blockIdx.x * 256 + threadIdx.x;
  if (i < n) out[i] = (bf16)in[i];
}

__global__ void add_inplace_kernel(float* __restrict__ x,
                                   const float* __restrict__ y, long long n) {
  long long i = (long long)blockIdx.x * 256 + threadIdx.x;
  if (i < n) x[i] += y[i];
}

// xs = 2*x + c1 + c2   (decoder: ln_torch input = x1 + x2)
__global__ void combo_kernel(float* __restrict__ o, const float* __restrict__ x,
                             const float* __restrict__ c1,
                             const float* __restrict__ c2, long long n) {
  long long i = (long long)blockIdx.x * 256 + threadIdx.x;
  if (i < n) o[i] = 2.0f * x[i] + c1[i] + c2[i];
}

__global__ void relu_cvt_kernel(const float* __restrict__ in,
                                bf16* __restrict__ out, long long n) {
  long long i = (long long)blockIdx.x * 256 + threadIdx.x;
  if (i < n) out[i] = (bf16)fmaxf(in[i], 0.0f);
}

// out[b,s,d] = emb[tok[b,s],d]*sqrt(D) + PE(s,d)
__global__ void embed_kernel(const int* __restrict__ tok,
                             const float* __restrict__ emb,
                             float* __restrict__ out) {
  const long long i = (long long)blockIdx.x * 256 + threadIdx.x;  // < M4*D
  if (i >= (long long)M4 * D) return;
  const int       d  = (int)(i & (D - 1));
  const long long bs = i >> 9;
  const int       s  = (int)(bs & (S - 1));
  const int       tk = tok[bs];
  const int       i2 = d >> 1;
  const float dv  = expf(-(float)(2 * i2) * 0.0179889460390160f); // ln(1e4)/512
  const float ang = (float)s * dv;
  const float pe  = (d & 1) ? cosf(ang) : sinf(ang);
  out[i] = emb[(long long)tk * D + d] * 22.6274169979695f + pe;   // sqrt(512)
}

// custom LN: unbiased std (ddof=1), eps added to std (1e-6)
__global__ void ln_pre_kernel(const float* __restrict__ x,
                              const float* __restrict__ ga,
                              const float* __restrict__ gb,
                              float* __restrict__ of, bf16* __restrict__ ob) {
  __shared__ float s1[256], s2[256];
  const int t = threadIdx.x;
  const float* xr = x + (long long)blockIdx.x * D;
  const float v0 = xr[t], v1 = xr[t + 256];
  s1[t] = v0 + v1; s2[t] = v0 * v0 + v1 * v1;
  __syncthreads();
  for (int s = 128; s > 0; s >>= 1) {
    if (t < s) { s1[t] += s1[t + s]; s2[t] += s2[t + s]; }
    __syncthreads();
  }
  const float mean = s1[0] * (1.0f / D);
  const float var  = fmaxf((s2[0] - (float)D * mean * mean) * (1.0f / (D - 1)), 0.0f);
  const float inv  = 1.0f / (sqrtf(var) + 1e-6f);
  const float y0 = ga[t] * (v0 - mean) * inv + gb[t];
  const float y1 = ga[t + 256] * (v1 - mean) * inv + gb[t + 256];
  if (of) { float* r = of + (long long)blockIdx.x * D; r[t] = y0; r[t + 256] = y1; }
  if (ob) { bf16*  r = ob + (long long)blockIdx.x * D; r[t] = (bf16)y0; r[t + 256] = (bf16)y1; }
}

// torch LN: biased var, eps inside sqrt (1e-5)
__global__ void ln_torch_kernel(const float* __restrict__ x,
                                const float* __restrict__ ga,
                                const float* __restrict__ gb,
                                float* __restrict__ of, bf16* __restrict__ ob) {
  __shared__ float s1[256], s2[256];
  const int t = threadIdx.x;
  const float* xr = x + (long long)blockIdx.x * D;
  const float v0 = xr[t], v1 = xr[t + 256];
  s1[t] = v0 + v1; s2[t] = v0 * v0 + v1 * v1;
  __syncthreads();
  for (int s = 128; s > 0; s >>= 1) {
    if (t < s) { s1[t] += s1[t + s]; s2[t] += s2[t + s]; }
    __syncthreads();
  }
  const float mean = s1[0] * (1.0f / D);
  const float var  = fmaxf(s2[0] * (1.0f / D) - mean * mean, 0.0f);
  const float inv  = rsqrtf(var + 1e-5f);
  const float y0 = (v0 - mean) * inv * ga[t] + gb[t];
  const float y1 = (v1 - mean) * inv * ga[t + 256] + gb[t + 256];
  if (of) { float* r = of + (long long)blockIdx.x * D; r[t] = y0; r[t + 256] = y1; }
  if (ob) { bf16*  r = ob + (long long)blockIdx.x * D; r[t] = (bf16)y0; r[t + 256] = (bf16)y1; }
}

// fp32 [B,S,H*DK] -> bf16 [B,H,S,DK]
__global__ void split_heads_kernel(const float* __restrict__ in,
                                   bf16* __restrict__ out) {
  const long long o = (long long)blockIdx.x * 256 + threadIdx.x;
  if (o >= (long long)M4 * D) return;
  const int dk = (int)(o & (DK - 1));
  const int s  = (int)((o >> 6) & (S - 1));
  const int h  = (int)((o >> 15) & (H - 1));
  const int b  = (int)(o >> 18);
  out[o] = (bf16)in[((long long)b * S + s) * D + h * DK + dk];
}

// fp32 [B,S,H*DK] -> bf16 [B,H,DK,S]   (V stored head-transposed for attn@V)
__global__ void split_headsT_kernel(const float* __restrict__ in,
                                    bf16* __restrict__ out) {
  const long long o = (long long)blockIdx.x * 256 + threadIdx.x;
  if (o >= (long long)M4 * D) return;
  const int s  = (int)(o & (S - 1));
  const int dk = (int)((o >> 9) & (DK - 1));
  const int h  = (int)((o >> 15) & (H - 1));
  const int b  = (int)(o >> 18);
  out[o] = (bf16)in[((long long)b * S + s) * D + h * DK + dk];
}

// fp32 [B,H,S,DK] -> bf16 [B,S,D]
__global__ void merge_heads_kernel(const float* __restrict__ in,
                                   bf16* __restrict__ out) {
  const long long o = (long long)blockIdx.x * 256 + threadIdx.x;
  if (o >= (long long)M4 * D) return;
  const int d = (int)(o & (D - 1));
  const int s = (int)((o >> 9) & (S - 1));
  const int b = (int)(o >> 18);
  const int h = d >> 6, dk = d & (DK - 1);
  out[o] = (bf16)in[(((long long)b * H + h) * S + s) * DK + dk];
}

// masked, scaled softmax over rows of length S; emits bf16 attn weights
__global__ void softmax_kernel(const float* __restrict__ sc,
                               const int* __restrict__ mask,
                               bf16* __restrict__ attn,
                               long long mask_b, long long mask_q) {
  __shared__ float red[256];
  const int t = threadIdx.x;
  const long long row = blockIdx.x;          // (b*H + h)*S + q
  const int q = (int)(row & (S - 1));
  const int b = (int)(row >> 12);            // / (H*S)
  const float* sr = sc + row * S;
  const int*   mr = mask + (long long)b * mask_b + (long long)q * mask_q;
  const float scale = 0.125f;                // 1/sqrt(DK)
  float v0 = (mr[t]       == 0) ? -1e9f : sr[t]       * scale;
  float v1 = (mr[t + 256] == 0) ? -1e9f : sr[t + 256] * scale;
  red[t] = fmaxf(v0, v1); __syncthreads();
  for (int s = 128; s > 0; s >>= 1) {
    if (t < s) red[t] = fmaxf(red[t], red[t + s]);
    __syncthreads();
  }
  const float m = red[0]; __syncthreads();
  const float e0 = expf(v0 - m), e1 = expf(v1 - m);
  red[t] = e0 + e1; __syncthreads();
  for (int s = 128; s > 0; s >>= 1) {
    if (t < s) red[t] += red[t + s];
    __syncthreads();
  }
  const float inv = 1.0f / red[0];
  bf16* ar = attn + row * S;
  ar[t] = (bf16)(e0 * inv); ar[t + 256] = (bf16)(e1 * inv);
}

// in-place log_softmax over rows of length V (runs directly on d_out)
__global__ void log_softmax_kernel(float* __restrict__ x) {
  __shared__ float red[256];
  const int t = threadIdx.x;
  float* xr = x + (long long)blockIdx.x * V;
  float mx = -3.4e38f;
  for (int i = t; i < V; i += 256) mx = fmaxf(mx, xr[i]);
  red[t] = mx; __syncthreads();
  for (int s = 128; s > 0; s >>= 1) {
    if (t < s) red[t] = fmaxf(red[t], red[t + s]);
    __syncthreads();
  }
  mx = red[0]; __syncthreads();
  float sum = 0.0f;
  for (int i = t; i < V; i += 256) sum += expf(xr[i] - mx);
  red[t] = sum; __syncthreads();
  for (int s = 128; s > 0; s >>= 1) {
    if (t < s) red[t] += red[t + s];
    __syncthreads();
  }
  const float lse = mx + logf(red[0]);
  for (int i = t; i < V; i += 256) xr[i] -= lse;
}

// ============================================================================
// Host orchestration
// ============================================================================
extern "C" void kernel_launch(void* const* d_in, const int* in_sizes, int n_in,
                              void* d_out, int out_size, void* d_ws, size_t ws_size,
                              hipStream_t stream) {
  (void)in_sizes; (void)out_size; (void)ws_size;

  // ---- input leaves (jax tree flatten: dict keys sorted at every level) ----
  const int* src       = (const int*)d_in[0];
  const int* src2      = (const int*)d_in[1];
  const int* tgt       = (const int*)d_in[2];
  const int* src_mask  = (const int*)d_in[3];
  const int* src2_mask = (const int*)d_in[4];
  const int* tgt_mask  = (const int*)d_in[5];
  auto P = [&](int i) -> const float* { return (const float*)d_in[i]; };

  // params order: dec (layers[4]{ff(w1:b,w; w2:b,w), lay(a,b), n[4](a,b),
  //   self,src1,src2 (k:b,w; o:b,w; q:b,w; v:b,w)}, norm(a,b)),
  //   emb1, emb2, emb3, enc1 (layers[3]{attn, ff, n1, n2}, norm), enc2, gen(b,w)
  const int PB = 6;
  const int DEC_NA = PB + 152, DEC_NB = PB + 153;
  const int EMB1 = PB + 154, EMB2 = PB + 155, EMB3 = PB + 156;
  const int E1 = PB + 157, E1NA = E1 + 48, E1NB = E1 + 49;
  const int E2 = PB + 207, E2NA = E2 + 48, E2NB = E2 + 49;
  const int GEN_B = PB + 257, GEN_W = PB + 258;
  if (n_in < GEN_W + 1) return;  // flattening assumption violated; bail safely

  // ---- workspace bump allocator ----
  char*  wsp = (char*)d_ws;
  size_t off = 0;
  auto alloc = [&](size_t bytes) -> void* {
    size_t a = (off + 255) & ~(size_t)255;
    off = a + bytes;
    return (void*)(wsp + a);
  };
  auto cvtW = [&](int leaf, size_t n) -> const bf16* {
    bf16* p = (bf16*)alloc(n * sizeof(bf16));
    cvt_f32_bf16_kernel<<<dim3((unsigned)((n + 255) / 256)), dim3(256), 0, stream>>>(
        P(leaf), p, (long long)n);
    return p;
  };

  struct MhaW { const float *bq, *bk, *bv, *bo; const bf16 *wq, *wk, *wv, *wo; };
  struct FfnW { const float *b1, *b2; const bf16 *w1, *w2; };
  auto mkMha = [&](int b0) -> MhaW {
    MhaW w;
    w.bk = P(b0 + 0); w.wk = cvtW(b0 + 1, (size_t)D * D);
    w.bo = P(b0 + 2); w.wo = cvtW(b0 + 3, (size_t)D * D);
    w.bq = P(b0 + 4); w.wq = cvtW(b0 + 5, (size_t)D * D);
    w.bv = P(b0 + 6); w.wv = cvtW(b0 + 7, (size_t)D * D);
    return w;
  };
  auto mkFfn = [&](int b0) -> FfnW {
    FfnW f;
    f.b1 = P(b0 + 0); f.w1 = cvtW(b0 + 1, (size_t)FF * D);
    f.b2 = P(b0 + 2); f.w2 = cvtW(b0 + 3, (size_t)D * FF);
    return f;
  };

  struct EncL { MhaW attn; FfnW ff; const float *n1a, *n1b, *n2a, *n2b; };
  struct DecL { MhaW self, s1m, s2m; FfnW ff; const float *laya, *layb;
                const float *na[4], *nb[4]; };

  DecL dec[4];
  for (int l = 0; l < 4; ++l) {
    const int d0 = PB + l * 38;
    dec[l].ff = mkFfn(d0 + 0);
    dec[l].laya = P(d0 + 4); dec[l].layb = P(d0 + 5);
    for (int i = 0; i < 4; ++i) { dec[l].na[i] = P(d0 + 6 + 2 * i); dec[l].nb[i] = P(d0 + 7 + 2 * i); }
    dec[l].self = mkMha(d0 + 14);
    dec[l].s1m  = mkMha(d0 + 22);
    dec[l].s2m  = mkMha(d0 + 30);
  }
  EncL enc1[3], enc2[3];
  for (int l = 0; l < 3; ++l) {
    const int e = E1 + l * 16;
    enc1[l].attn = mkMha(e + 0); enc1[l].ff = mkFfn(e + 8);
    enc1[l].n1a = P(e + 12); enc1[l].n1b = P(e + 13);
    enc1[l].n2a = P(e + 14); enc1[l].n2b = P(e + 15);
  }
  for (int l = 0; l < 3; ++l) {
    const int e = E2 + l * 16;
    enc2[l].attn = mkMha(e + 0); enc2[l].ff = mkFfn(e + 8);
    enc2[l].n1a = P(e + 12); enc2[l].n1b = P(e + 13);
    enc2[l].n2a = P(e + 14); enc2[l].n2b = P(e + 15);
  }
  const float* genB = P(GEN_B);
  const bf16*  genW = cvtW(GEN_W, (size_t)V * D);

  // ---- activation scratch ----
  float* xbuf   = (float*)alloc((size_t)M4 * D * 4);   // residual stream (fp32)
  bf16*  ybf    = (bf16*) alloc((size_t)M4 * D * 2);   // LN output (GEMM A)
  float* tmpf   = (float*)alloc((size_t)M4 * D * 4);   // projection out
  bf16*  Qh     = (bf16*) alloc((size_t)BH * S * DK * 2);
  bf16*  Kh     = (bf16*) alloc((size_t)BH * S * DK * 2);
  bf16*  Vt     = (bf16*) alloc((size_t)BH * DK * S * 2);
  float* scores = (float*)alloc((size_t)BH * S * S * 4);  // 64MB
  bf16*  attnb  = (bf16*) alloc((size_t)BH * S * S * 2);  // 32MB
  float* Oh     = (float*)alloc((size_t)BH * S * DK * 4);
  bf16*  mrg    = (bf16*) alloc((size_t)M4 * D * 2);
  float* c1     = (float*)alloc((size_t)M4 * D * 4);
  float* c2     = (float*)alloc((size_t)M4 * D * 4);
  float* xs     = (float*)alloc((size_t)M4 * D * 4);
  bf16*  m1b    = (bf16*) alloc((size_t)M4 * D * 2);
  bf16*  m2b    = (bf16*) alloc((size_t)M4 * D * 2);
  bf16*  decb   = (bf16*) alloc((size_t)M4 * D * 2);
  // FFN hidden aliases the (then-dead) attention score buffers
  float* hid = scores;                 // [M4,FF] fp32 (32MB <= 64MB)
  bf16*  hbf = attnb;                  // [M4,FF] bf16 (16MB <= 32MB)

  const long long ND = (long long)M4 * D;
  const unsigned  GD = (unsigned)(ND / 256);  // 8192

  // ---- launch helpers ----
  auto gemm244 = [&](const bf16* A, const bf16* W, const float* bias, float* C,
                     int M, int N, int K, long long sA, long long sW,
                     long long sC, int batch) {
    gemm_bf16_wmma<2, 4, 4><<<dim3(N / 256, M / 32, batch), dim3(256), 0, stream>>>(
        A, W, bias, C, M, N, K, sA, sW, sC);
  };
  auto gemm814 = [&](const bf16* A, const bf16* W, const float* bias, float* C,
                     int M, int N, int K, long long sA, long long sW,
                     long long sC, int batch) {
    gemm_bf16_wmma<8, 1, 4><<<dim3(N / 64, M / 128, batch), dim3(256), 0, stream>>>(
        A, W, bias, C, M, N, K, sA, sW, sC);
  };
  auto ln_pre = [&](const float* x, const float* a, const float* b,
                    float* of, bf16* ob) {
    ln_pre_kernel<<<dim3(M4), dim3(256), 0, stream>>>(x, a, b, of, ob);
  };
  auto add_ip = [&](float* x, const float* y) {
    add_inplace_kernel<<<dim3(GD), dim3(256), 0, stream>>>(x, y, ND);
  };

  auto run_mha = [&](const bf16* qin, const bf16* kvin, const MhaW& w,
                     const int* mask, long long mb, long long mq, float* out) {
    gemm244(qin, w.wq, w.bq, tmpf, M4, D, D, 0, 0, 0, 1);
    split_heads_kernel<<<dim3(GD), dim3(256), 0, stream>>>(tmpf, Qh);
    gemm244(kvin, w.wk, w.bk, tmpf, M4, D, D, 0, 0, 0, 1);
    split_heads_kernel<<<dim3(GD), dim3(256), 0, stream>>>(tmpf, Kh);
    gemm244(kvin, w.wv, w.bv, tmpf, M4, D, D, 0, 0, 0, 1);
    split_headsT_kernel<<<dim3(GD), dim3(256), 0, stream>>>(tmpf, Vt);
    // scores[bh] = Qh[bh] @ Kh[bh]^T      (M=S, N=S, K=DK)
    gemm244(Qh, Kh, nullptr, scores, S, S, DK,
            (long long)S * DK, (long long)S * DK, (long long)S * S, BH);
    softmax_kernel<<<dim3(BH * S), dim3(256), 0, stream>>>(scores, mask, attnb, mb, mq);
    // Oh[bh] = attn[bh] @ Vt[bh]^T        (M=S, N=DK, K=S)
    gemm814(attnb, Vt, nullptr, Oh, S, DK, S,
            (long long)S * S, (long long)DK * S, (long long)S * DK, BH);
    merge_heads_kernel<<<dim3(GD), dim3(256), 0, stream>>>(Oh, mrg);
    gemm244(mrg, w.wo, w.bo, out, M4, D, D, 0, 0, 0, 1);
  };

  auto run_ffn = [&](const bf16* yin, const FfnW& w, float* out) {
    gemm244(yin, w.w1, w.b1, hid, M4, FF, D, 0, 0, 0, 1);
    relu_cvt_kernel<<<dim3((unsigned)((long long)M4 * FF / 256)), dim3(256), 0,
                     stream>>>(hid, hbf, (long long)M4 * FF);
    gemm244(hbf, w.w2, w.b2, out, M4, D, FF, 0, 0, 0, 1);
  };

  auto run_encoder = [&](const int* tokens, const float* embp, EncL* Ls,
                         const float* na, const float* nb, const int* mask,
                         bf16* memb) {
    embed_kernel<<<dim3(GD), dim3(256), 0, stream>>>(tokens, embp, xbuf);
    for (int l = 0; l < 3; ++l) {
      ln_pre(xbuf, Ls[l].n1a, Ls[l].n1b, nullptr, ybf);
      run_mha(ybf, ybf, Ls[l].attn, mask, (long long)S, 0, c1);
      add_ip(xbuf, c1);
      ln_pre(xbuf, Ls[l].n2a, Ls[l].n2b, nullptr, ybf);
      run_ffn(ybf, Ls[l].ff, c1);
      add_ip(xbuf, c1);
    }
    ln_pre(xbuf, na, nb, nullptr, memb);  // memory only consumed as bf16 GEMM input
  };

  // ---- forward pass ----
  run_encoder(src,  P(EMB1), enc1, P(E1NA), P(E1NB), src_mask,  m1b);
  run_encoder(src2, P(EMB2), enc2, P(E2NA), P(E2NB), src2_mask, m2b);

  embed_kernel<<<dim3(GD), dim3(256), 0, stream>>>(tgt, P(EMB3), xbuf);
  for (int l = 0; l < 4; ++l) {
    DecL& L = dec[l];
    ln_pre(xbuf, L.na[0], L.nb[0], nullptr, ybf);
    run_mha(ybf, ybf, L.self, tgt_mask, (long long)Tt * Tt, (long long)Tt, c1);
    add_ip(xbuf, c1);
    ln_pre(xbuf, L.na[1], L.nb[1], nullptr, ybf);
    run_mha(ybf, m1b, L.s1m, src_mask, (long long)S, 0, c1);
    ln_pre(xbuf, L.na[2], L.nb[2], nullptr, ybf);
    run_mha(ybf, m2b, L.s2m, src2_mask, (long long)S, 0, c2);
    combo_kernel<<<dim3(GD), dim3(256), 0, stream>>>(xs, xbuf, c1, c2, ND);
    ln_torch_kernel<<<dim3(M4), dim3(256), 0, stream>>>(xs, L.laya, L.layb, xbuf, nullptr);
    ln_pre(xbuf, L.na[3], L.nb[3], nullptr, ybf);
    run_ffn(ybf, L.ff, c1);
    add_ip(xbuf, c1);
  }
  ln_pre(xbuf, P(DEC_NA), P(DEC_NB), nullptr, decb);

  // generator: logits straight into d_out, then in-place log_softmax
  gemm244(decb, genW, genB, (float*)d_out, M4, V, D, 0, 0, 0, 1);
  log_softmax_kernel<<<dim3(M4), dim3(256), 0, stream>>>((float*)d_out);
}
